// MultiHeadSelfAttention3D_42253888258489
// MI455X (gfx1250) — matrix operand order repaired and checked by reading, not compiled, not run on hardware
//
#include <hip/hip_runtime.h>
#include <hip/hip_bf16.h>

typedef __bf16 bf16_t;
typedef __attribute__((ext_vector_type(16))) __bf16        v16bf;
typedef __attribute__((ext_vector_type(8)))  float         v8f;
typedef __attribute__((ext_vector_type(8)))  unsigned int  v8u;

#define C_DIM   256
#define N_DIM   4096
#define NH      8
#define HD      32

// ---------------------------------------------------------------------------
// WMMA tile loaders (wave32, 16-bit operands); this code uses contiguous k order k = 16*(lane>>4) + s in A and B.
// A 16x32 (MxK), row-major source: row i = lane&15; VGPR r holds K pair
//   kb = 2r+16*kg, kg = lane>>4. kb even ->
//   each VGPR is one aligned dword load of two consecutive bf16.
__device__ __forceinline__ v16bf load_tileA_pairs(const bf16_t* base, int stride, int lane) {
  int i  = lane & 15;
  int kg = (lane >> 4) & 1;
  v8u u;
#pragma unroll
  for (int r = 0; r < 8; ++r) {
    int kb = 2 * r + 16 * kg;
    u[r] = *reinterpret_cast<const unsigned int*>(base + i * stride + kb);
  }
  return __builtin_bit_cast(v16bf, u);
}

// B 32x16 (KxN) from COLUMN-major storage (element (k,j) at base[j*stride+k]):
//   col j = lane&15; VGPR r holds K pair (2r+16*kh, +1), kh = lane>>4.
__device__ __forceinline__ v16bf load_tileB_pairsT(const bf16_t* base, int stride, int lane) {
  int j  = lane & 15;
  int kh = (lane >> 4) & 1;
  v8u u;
#pragma unroll
  for (int r = 0; r < 8; ++r) {
    int kb = 2 * r + 16 * kh;
    u[r] = *reinterpret_cast<const unsigned int*>(base + j * stride + kb);
  }
  return __builtin_bit_cast(v16bf, u);
}
// C/D f32 16x16: VGPR r holds row (r + 8*(lane>>4)), col = lane&15.

__device__ __forceinline__ unsigned int pack_bf16(float a, float b) {
  unsigned short ua = __builtin_bit_cast(unsigned short, (bf16_t)a);
  unsigned short ub = __builtin_bit_cast(unsigned short, (bf16_t)b);
  return (unsigned int)ua | ((unsigned int)ub << 16);
}

// 64x16 strip GEMM core, software-pipelined (double-buffered tiles).
// K must be a power of two; last-iteration prefetch wraps to k=0 (dead).
__device__ __forceinline__ void gemm_core_64x16(const bf16_t* wrow, const bf16_t* xcol,
                                                int K, int lane, v8f acc[4]) {
  v16bf bcur = load_tileB_pairsT(xcol, K, lane);
  v16bf acur[4];
#pragma unroll
  for (int t = 0; t < 4; ++t) acur[t] = load_tileA_pairs(wrow + t * 16 * K, K, lane);

  for (int k0 = 0; k0 < K; k0 += 32) {
    int kn = (k0 + 32) & (K - 1);
    __builtin_prefetch(wrow + k0 + 64, 0, 0);
    v16bf bnxt = load_tileB_pairsT(xcol + kn, K, lane);
    v16bf anxt[4];
#pragma unroll
    for (int t = 0; t < 4; ++t) anxt[t] = load_tileA_pairs(wrow + t * 16 * K + kn, K, lane);
#pragma unroll
    for (int t = 0; t < 4; ++t)
      acc[t] = __builtin_amdgcn_wmma_f32_16x16x32_bf16(false, acur[t], false, bcur,
                                                       (short)0, acc[t], false, false);
    bcur = bnxt;
#pragma unroll
    for (int t = 0; t < 4; ++t) acur[t] = anxt[t];
  }
}

// ---------------------------------------------------------------------------
__global__ void cvt_f32_to_bf16_kernel(const float* __restrict__ src,
                                       bf16_t* __restrict__ dst, int n) {
  int idx = blockIdx.x * blockDim.x + threadIdx.x;
  if (idx < n) dst[idx] = (bf16_t)src[idx];
}

// x (C x N row-major f32) -> xT (N x C bf16); coalesced writes
__global__ void cvt_x_transpose_kernel(const float* __restrict__ src,
                                       bf16_t* __restrict__ dst) {
  int idx = blockIdx.x * blockDim.x + threadIdx.x;   // idx = n*C + c
  int n = idx >> 8;
  int c = idx & (C_DIM - 1);
  dst[idx] = (bf16_t)src[c * N_DIM + n];
}

// ---------------------------------------------------------------------------
// qkv = qkv_w(768x256) @ x(256x4096). Each wave: 64-row x 16-col strip.
//   Q -> qtb[h][n][d], K -> ktb[h][m][d] (packed dword stores)
//   V -> vb row-major (A-operand of P*V needs m contiguous)
__global__ void gemm_qkv_kernel(const bf16_t* __restrict__ W,
                                const bf16_t* __restrict__ xT,
                                bf16_t* __restrict__ qtb,
                                bf16_t* __restrict__ ktb,
                                bf16_t* __restrict__ vb) {
  const int K = C_DIM, Nn = N_DIM;
  int lane = threadIdx.x & 31;
  int wave = __builtin_amdgcn_readfirstlane(threadIdx.x >> 5);   // scalar
  int tile = blockIdx.x * (blockDim.x >> 5) + wave;
  int tiles_n = Nn >> 4;                         // 256
  int tm = tile / tiles_n;                       // 0..11 (64-row blocks)
  int tn = tile - tm * tiles_n;

  const bf16_t* wrow = W + tm * 64 * K;
  const bf16_t* xcol = xT + (tn * 16) * K;

  v8f acc[4] = {};
  gemm_core_64x16(wrow, xcol, K, lane, acc);

  int j  = lane & 15;
  int hi = (lane >> 4) & 1;
  int c  = tn * 16 + j;
#pragma unroll
  for (int t = 0; t < 4; ++t) {
    int row0 = tm * 64 + t * 16;                 // scalar
    int seg  = row0 >> 8;                        // 0=Q, 1=K, 2=V (scalar branch)
    if (seg == 2) {                              // V rows, row-major store
#pragma unroll
      for (int r = 0; r < 8; ++r)
        vb[(row0 + r + 8 * hi - 2 * C_DIM) * Nn + c] = (bf16_t)acc[t][r];
    } else {                                     // Q/K: [h][n|m][d] packed
      int rr = row0 - seg * C_DIM + 8 * hi;      // d runs along r (contiguous)
      int h  = rr >> 5, dbase = rr & 31;
      bf16_t* dstb = (seg == 0 ? qtb : ktb) + h * (N_DIM * HD) + c * HD + dbase;
#pragma unroll
      for (int u = 0; u < 4; ++u)
        *reinterpret_cast<unsigned int*>(dstb + 2 * u) =
            pack_bf16(acc[t][2 * u], acc[t][2 * u + 1]);
    }
  }
}

// ---------------------------------------------------------------------------
// Flash attention, S^T orientation: St = K^T(16m x 32d) * Q(32d x 16n).
// Online softmax state is one scalar per lane; one shfl_xor(16) per chunk.
// O = V(16d x 32m) * P^T; P^T D->B relayout is a half-wave swap of packed
// bf16 dwords (no LDS). K/V tiles double-buffered across the m-loop.
__global__ void attn_flash_kernel(const bf16_t* __restrict__ qtb,
                                  const bf16_t* __restrict__ ktb,
                                  const bf16_t* __restrict__ vb,
                                  bf16_t* __restrict__ aoT) {
  int lane = threadIdx.x & 31;
  int wave = __builtin_amdgcn_readfirstlane(threadIdx.x >> 5);
  int gw   = blockIdx.x * (blockDim.x >> 5) + wave;
  int h     = gw >> 8;                        // 256 n-tiles per head
  int ntile = gw & 255;
  int n0    = ntile * 16;
  int j  = lane & 15;
  int hi = (lane >> 4) & 1;

  // Q B-tile (32d x 16n) from n-major storage, loaded once per wave
  v16bf bq = load_tileB_pairsT(qtb + h * (N_DIM * HD) + n0 * HD, HD, lane);

  const bf16_t* kt  = ktb + h * (N_DIM * HD);      // [m][d], m-major
  const bf16_t* vh0 = vb + (h * HD) * N_DIM;       // d = 0..15
  const bf16_t* vh1 = vb + (h * HD + 16) * N_DIM;  // d = 16..31

  const float scale = 0.1767766952966369f;    // 32^-0.5
  float mrun = -__builtin_inff();
  float lrun = 0.0f;
  v8f o0 = {}, o1 = {};

  v16bf ak0c = load_tileA_pairs(kt, HD, lane);
  v16bf ak1c = load_tileA_pairs(kt + 16 * HD, HD, lane);
  v16bf av0c = load_tileA_pairs(vh0, N_DIM, lane);
  v16bf av1c = load_tileA_pairs(vh1, N_DIM, lane);

  for (int m0 = 0; m0 < N_DIM; m0 += 32) {
    int mn = (m0 + 32) & (N_DIM - 1);         // wraps to 0 on last iter (dead)
    __builtin_prefetch(kt + (m0 + 64) * HD, 0, 0);
    __builtin_prefetch(vh0 + m0 + 64, 0, 0);
    __builtin_prefetch(vh1 + m0 + 64, 0, 0);
    v16bf ak0n = load_tileA_pairs(kt + mn * HD, HD, lane);
    v16bf ak1n = load_tileA_pairs(kt + (mn + 16) * HD, HD, lane);
    v16bf av0n = load_tileA_pairs(vh0 + mn, N_DIM, lane);
    v16bf av1n = load_tileA_pairs(vh1 + mn, N_DIM, lane);

    // St0: m0..m0+15, St1: m0+16..m0+31 (raw logits; scale folded into exp)
    v8f z = {};
    v8f s0 = __builtin_amdgcn_wmma_f32_16x16x32_bf16(false, ak0c, false, bq,
                                                     (short)0, z, false, false);
    v8f s1 = __builtin_amdgcn_wmma_f32_16x16x32_bf16(false, ak1c, false, bq,
                                                     (short)0, z, false, false);

    float cm = fmaxf(s0[0], s1[0]);
#pragma unroll
    for (int r = 1; r < 8; ++r) cm = fmaxf(cm, fmaxf(s0[r], s1[r]));
    cm = fmaxf(cm, __shfl_xor(cm, 16, 32));
    float mnew  = fmaxf(mrun, cm);
    float alpha = __expf((mrun - mnew) * scale);    // 0 on first chunk
    mrun = mnew;

    float p0[8], p1[8], ls = 0.0f;
#pragma unroll
    for (int r = 0; r < 8; ++r) {
      p0[r] = __expf((s0[r] - mnew) * scale);
      p1[r] = __expf((s1[r] - mnew) * scale);
      ls += p0[r] + p1[r];
    }
    lrun = lrun * alpha + ls;
#pragma unroll
    for (int r = 0; r < 8; ++r) { o0[r] *= alpha; o1[r] *= alpha; }

    // P^T D-layout -> B-layout: half-wave swap of packed pairs
    v8u bu;
#pragma unroll
    for (int u = 0; u < 4; ++u) {
      unsigned int A4 = pack_bf16(p0[2 * u], p0[2 * u + 1]);
      unsigned int B4 = pack_bf16(p1[2 * u], p1[2 * u + 1]);
      unsigned int xA = (unsigned int)__shfl_xor((int)A4, 16, 32);
      unsigned int xB = (unsigned int)__shfl_xor((int)B4, 16, 32);
      bu[u]     = hi ? xB : A4;
      bu[u + 4] = hi ? B4 : xA;
    }
    v16bf bp = __builtin_bit_cast(v16bf, bu);

    o0 = __builtin_amdgcn_wmma_f32_16x16x32_bf16(false, av0c, false, bp,
                                                 (short)0, o0, false, false);
    o1 = __builtin_amdgcn_wmma_f32_16x16x32_bf16(false, av1c, false, bp,
                                                 (short)0, o1, false, false);
    ak0c = ak0n; ak1c = ak1n; av0c = av0n; av1c = av1n;
  }

  lrun += __shfl_xor(lrun, 16, 32);
  float linv = 1.0f / lrun;

  // aoT[n][c]: d contiguous along r -> packed dword stores
  bf16_t* orow = aoT + (n0 + j) * C_DIM + h * HD + 8 * hi;
#pragma unroll
  for (int u = 0; u < 4; ++u) {
    *reinterpret_cast<unsigned int*>(orow + 2 * u) =
        pack_bf16(o0[2 * u] * linv, o0[2 * u + 1] * linv);
    *reinterpret_cast<unsigned int*>(orow + 16 + 2 * u) =
        pack_bf16(o1[2 * u] * linv, o1[2 * u + 1] * linv);
  }
}

// ---------------------------------------------------------------------------
// out(256x4096 f32) = proj_w(256x256) @ att(256x4096) + bias; att read from
// transposed aoT (n-major) so B tiles are pair-loads. 64x16 strip per wave.
__global__ void gemm_proj_kernel(const bf16_t* __restrict__ W,
                                 const bf16_t* __restrict__ aoT,
                                 const float* __restrict__ bias,
                                 float* __restrict__ Y) {
  const int K = C_DIM, Nn = N_DIM;
  int lane = threadIdx.x & 31;
  int wave = __builtin_amdgcn_readfirstlane(threadIdx.x >> 5);
  int tile = blockIdx.x * (blockDim.x >> 5) + wave;
  int tiles_n = Nn >> 4;
  int tm = tile / tiles_n;                       // 0..3
  int tn = tile - tm * tiles_n;

  const bf16_t* wrow = W + tm * 64 * K;
  const bf16_t* xcol = aoT + (tn * 16) * K;

  v8f acc[4] = {};
  gemm_core_64x16(wrow, xcol, K, lane, acc);

  int j  = lane & 15;
  int hi = (lane >> 4) & 1;
#pragma unroll
  for (int t = 0; t < 4; ++t) {
#pragma unroll
    for (int r = 0; r < 8; ++r) {
      int row = tm * 64 + t * 16 + r + 8 * hi;
      Y[row * Nn + tn * 16 + j] = acc[t][r] + bias[row];
    }
  }
}

// ---------------------------------------------------------------------------
extern "C" void kernel_launch(void* const* d_in, const int* in_sizes, int n_in,
                              void* d_out, int out_size, void* d_ws, size_t ws_size,
                              hipStream_t stream) {
  const float* x      = (const float*)d_in[0];   // 1 x 256 x 16 x 16 x 16
  const float* qkv_w  = (const float*)d_in[1];   // 768 x 256
  const float* proj_w = (const float*)d_in[2];   // 256 x 256
  const float* proj_b = (const float*)d_in[3];   // 256

  char* ws = (char*)d_ws;
  bf16_t* xT  = (bf16_t*)(ws);                                    // 4096x256 (2 MB)
  bf16_t* wq  = (bf16_t*)(ws + (size_t)2 * 1024 * 1024);          // 768x256
  bf16_t* wp  = (bf16_t*)(ws + (size_t)2 * 1024 * 1024 + 512 * 1024); // 256x256
  bf16_t* qtb = (bf16_t*)(ws + (size_t)3 * 1024 * 1024);          // Q^T 8x4096x32
  bf16_t* ktb = (bf16_t*)(ws + (size_t)5 * 1024 * 1024);          // K^T 8x4096x32
  bf16_t* vb  = (bf16_t*)(ws + (size_t)7 * 1024 * 1024);          // V 256x4096
  bf16_t* aoT = (bf16_t*)(ws + (size_t)9 * 1024 * 1024);          // att^T 4096x256

  int nq = 3 * C_DIM * C_DIM;
  int np = C_DIM * C_DIM;
  cvt_x_transpose_kernel<<<(C_DIM * N_DIM) / 256, 256, 0, stream>>>(x, xT);
  cvt_f32_to_bf16_kernel<<<(nq + 255) / 256, 256, 0, stream>>>(qkv_w, wq, nq);
  cvt_f32_to_bf16_kernel<<<(np + 255) / 256, 256, 0, stream>>>(proj_w, wp, np);

  // 12 m-blocks * 256 n-tiles = 3072 waves, 8 waves/block
  gemm_qkv_kernel<<<3072 / 8, 256, 0, stream>>>(wq, xT, qtb, ktb, vb);

  // 8 heads * 256 n-tiles = 2048 waves
  attn_flash_kernel<<<2048 / 8, 256, 0, stream>>>(qtb, ktb, vb, aoT);

  // 4 m-blocks * 256 n-tiles = 1024 waves
  gemm_proj_kernel<<<1024 / 8, 256, 0, stream>>>(wp, aoT, proj_b, (float*)d_out);
}
